// RegionTokenBuilder_67113158968010
// MI455X (gfx1250) — matrix operand order repaired and checked by reading, not compiled, not run on hardware
//
#include <hip/hip_runtime.h>
#include <hip/hip_bf16.h>

typedef __attribute__((ext_vector_type(16))) __bf16 v16bf;
typedef __attribute__((ext_vector_type(8)))  float  v8f;

constexpr int NB = 32;     // batch
constexpr int NS = 2048;   // seq len (== max regions, padded)
constexpr int ND = 768;    // dim
constexpr int NTOK = 256;  // token vocab for tokens_level

// -------------------------------------------------------------------------
// Kernel 1: per-batch region metadata.
//   - valid_len from prefix mask
//   - first-occurrence position per token value (256-entry LDS table)
//   - is_first flags + block-wide prefix sum -> ranks
//   - region_ids scatter, step_to_region, region_mask, mlm_mask, counts
// Writes float views into d_out and int/float working copies into d_ws.
// -------------------------------------------------------------------------
__global__ __launch_bounds__(256) void region_meta_kernel(
    const int* __restrict__ tokens, const int* __restrict__ amask,
    float* __restrict__ out, int* __restrict__ s2r_i, int* __restrict__ rid_i,
    float* __restrict__ cnt_f)
{
  const int b = blockIdx.x;
  const int t = threadIdx.x;

  __shared__ int firstpos[NTOK];
  __shared__ int cnts[NTOK];
  __shared__ int regOfTok[NTOK];
  __shared__ int rankArr[NS];
  __shared__ int scanbuf[256];
  __shared__ int lenSh;

  const int* tok = tokens + b * NS;
  const int* msk = amask  + b * NS;

  firstpos[t] = 0x7fffffff;
  cnts[t] = 0;
  if (t == 0) lenSh = 0;
  __syncthreads();

  // prefix mask -> length = popcount
  int lcl = 0;
  #pragma unroll
  for (int j = 0; j < 8; ++j) lcl += msk[t * 8 + j];
  atomicAdd(&lenSh, lcl);
  __syncthreads();
  const int len = lenSh;

  // first valid occurrence per token value
  #pragma unroll
  for (int j = 0; j < 8; ++j) {
    int s = t * 8 + j;
    if (s < len) atomicMin(&firstpos[tok[s]], s);
  }
  __syncthreads();

  // is_first flags + local sums
  int f[8]; int tot = 0;
  #pragma unroll
  for (int j = 0; j < 8; ++j) {
    int s = t * 8 + j;
    int isf = (s < len) && (firstpos[tok[s]] == s);
    f[j] = isf; tot += isf;
  }
  scanbuf[t] = tot;
  __syncthreads();
  // Hillis-Steele inclusive scan over 256 per-thread totals
  for (int off = 1; off < 256; off <<= 1) {
    int v = scanbuf[t];
    int add = (t >= off) ? scanbuf[t - off] : 0;
    __syncthreads();
    scanbuf[t] = v + add;
    __syncthreads();
  }
  const int nreg = scanbuf[255];
  int run = (t > 0) ? scanbuf[t - 1] : 0;
  #pragma unroll
  for (int j = 0; j < 8; ++j) {
    int s = t * 8 + j;
    run += f[j];
    rankArr[s] = run - 1;     // inclusive cumsum - 1
  }
  __syncthreads();

  // region index of each token value
  {
    int fp = firstpos[t];
    regOfTok[t] = (fp != 0x7fffffff) ? rankArr[fp] : -1;
  }
  __syncthreads();

  const size_t oRT = (size_t)NB * NS * ND;
  float* out_rid = out + oRT + (size_t)b * NS;
  float* out_msk = out + oRT + (size_t)NB * NS     + (size_t)b * NS;
  float* out_s2r = out + oRT + (size_t)2 * NB * NS + (size_t)b * NS;
  float* out_mlm = out + oRT + (size_t)3 * NB * NS + (size_t)b * NS;

  // zero-fill region_ids rows; masks
  for (int s = t; s < NS; s += 256) {
    rid_i[b * NS + s] = 0;
    out_rid[s] = 0.f;
    out_msk[s] = (s < nreg) ? 1.f : 0.f;
    out_mlm[s] = 0.f;
  }
  __syncthreads();

  // scatter ids at first occurrences; step_to_region; counts
  #pragma unroll
  for (int j = 0; j < 8; ++j) {
    int s = t * 8 + j;
    if (f[j]) {
      int r = rankArr[s];
      rid_i[b * NS + r] = tok[s];
      out_rid[r] = (float)tok[s];
    }
    int reg = (s < len) ? regOfTok[tok[s]] : -1;
    s2r_i[b * NS + s] = reg;
    out_s2r[s] = (float)reg;
    if (reg >= 0) atomicAdd(&cnts[reg], 1);
  }
  __syncthreads();
  cnt_f[b * NTOK + t] = (float)cnts[t];
}

// -------------------------------------------------------------------------
// Kernel 2: WMMA one-hot segment-sum + epilogue for regions [0,256).
// Wave = one 16-wide d-stripe; 16 accumulator tiles cover all 256 regions,
// so step_embed is streamed exactly once. bf16 hi/lo split: two
// v_wmma_f32_16x16x32_bf16 per active tile -> fp32-class accuracy.
// grid = (ND/128, NB), block = 256 (8 waves of 16 cols each).
// -------------------------------------------------------------------------
__global__ __launch_bounds__(256) void region_gemm_kernel(
    const float* __restrict__ se, const float* __restrict__ ew,
    const int* __restrict__ s2r_i, const int* __restrict__ rid_i,
    const float* __restrict__ cnt_f, float* __restrict__ out)
{
  const int b    = blockIdx.y;
  const int wave = threadIdx.x >> 5;
  const int lane = threadIdx.x & 31;
  const int d0   = (blockIdx.x * 8 + wave) * 16;
  const int g    = (lane >> 4) & 1;   // half-wave group
  const int n    = lane & 15;         // N column / M row within tile

  v8f acc[16];
  #pragma unroll
  for (int tt = 0; tt < 16; ++tt)
    #pragma unroll
    for (int i = 0; i < 8; ++i) acc[tt][i] = 0.f;

  const int*   s2r_b = s2r_i + b * NS;
  const float* se_b  = se + (size_t)b * NS * ND;

  for (int s0 = 0; s0 < NS; s0 += 32) {
    // region ids for my 16 A-matrix K slots (contiguous K order, same as B tile):
    //   e=0..15 -> K = 16g + e
    const int4* rp = (const int4*)(s2r_b + s0);
    const int b4 = g * 4;
    int4 q0 = rp[b4 + 0];
    int4 q1 = rp[b4 + 1];
    int4 q2 = rp[b4 + 2];
    int4 q3 = rp[b4 + 3];
    int rk[16] = {q0.x, q0.y, q0.z, q0.w, q1.x, q1.y, q1.z, q1.w,
                  q2.x, q2.y, q2.z, q2.w, q3.x, q3.y, q3.z, q3.w};

    // wave-uniform mask of touched region-tiles (keeps EXEC all-ones)
    unsigned tmv = 0;
    #pragma unroll
    for (int e = 0; e < 16; ++e)
      if (rk[e] >= 0) tmv |= 1u << (rk[e] >> 4);
    #pragma unroll
    for (int off = 16; off > 0; off >>= 1)
      tmv |= (unsigned)__shfl_xor((int)tmv, off, 32);
    const unsigned tm = (unsigned)__builtin_amdgcn_readfirstlane((int)tmv);
    if (!tm) continue;  // whole chunk invalid (past valid_len)

    // B tile (32x16): lane n = column, g selects K=0..15 / K=16..31
    v16bf bhi, blo;
    const float* eb = se_b + (size_t)(s0 + g * 16) * ND + d0 + n;
    #pragma unroll
    for (int e = 0; e < 16; ++e) {
      float x  = eb[(size_t)e * ND];
      __bf16 h = (__bf16)x;
      float hf = (float)h;
      __bf16 l = (__bf16)(x - hf);
      bhi[e] = h; blo[e] = l;
    }

    #pragma unroll
    for (int tt = 0; tt < 16; ++tt) {
      if (tm & (1u << tt)) {
        v16bf a;
        const int target = tt * 16 + n;  // one-hot row within tile
        #pragma unroll
        for (int e = 0; e < 16; ++e)
          a[e] = (rk[e] == target) ? (__bf16)1.0f : (__bf16)0.0f;
        acc[tt] = __builtin_amdgcn_wmma_f32_16x16x32_bf16(
            false, a, false, bhi, (short)0, acc[tt], false, false);
        acc[tt] = __builtin_amdgcn_wmma_f32_16x16x32_bf16(
            false, a, false, blo, (short)0, acc[tt], false, false);
      }
    }
  }

  // epilogue: mean + embedding gather, store region_tokens rows [0,256)
  #pragma unroll
  for (int tt = 0; tt < 16; ++tt) {
    #pragma unroll
    for (int i = 0; i < 8; ++i) {
      const int r = tt * 16 + g * 8 + i;  // C/D layout: vgpr i -> M=i(+8)
      const int d = d0 + n;
      float c  = cnt_f[b * NTOK + r];
      int   id = rid_i[b * NS + r];
      float val = acc[tt][i] / fmaxf(c, 1.f) + ew[(size_t)id * ND + d];
      out[((size_t)b * NS + r) * ND + d] = val;
    }
  }
}

// -------------------------------------------------------------------------
// Kernel 3: rows r in [256, NS) are exactly embed_weight[PAD_ID=0].
// Pure bandwidth: one block per row, float4 coalesced stores.
// -------------------------------------------------------------------------
__global__ __launch_bounds__(192) void region_tail_kernel(
    const float* __restrict__ ew, float* __restrict__ out)
{
  const int row = blockIdx.x;
  const int b = row / (NS - NTOK);
  const int r = NTOK + (row - b * (NS - NTOK));
  const float4* src = (const float4*)ew;  // row 0 of embed_weight
  float4* dst = (float4*)(out + ((size_t)b * NS + r) * ND);
  dst[threadIdx.x] = src[threadIdx.x];    // ND/4 = 192 float4 per row
}

extern "C" void kernel_launch(void* const* d_in, const int* in_sizes, int n_in,
                              void* d_out, int out_size, void* d_ws, size_t ws_size,
                              hipStream_t stream) {
  (void)in_sizes; (void)n_in; (void)out_size; (void)ws_size;
  const int*   tokens = (const int*)d_in[0];
  const float* se     = (const float*)d_in[1];
  const int*   amask  = (const int*)d_in[2];
  const float* ew     = (const float*)d_in[3];
  float* out = (float*)d_out;

  int*   s2r_i = (int*)d_ws;                 // [NB*NS] step_to_region
  int*   rid_i = s2r_i + NB * NS;            // [NB*NS] region_ids
  float* cnt_f = (float*)(rid_i + NB * NS);  // [NB*NTOK] counts

  hipLaunchKernelGGL(region_meta_kernel, dim3(NB), dim3(256), 0, stream,
                     tokens, amask, out, s2r_i, rid_i, cnt_f);
  hipLaunchKernelGGL(region_gemm_kernel, dim3(ND / 128, NB), dim3(256), 0, stream,
                     se, ew, s2r_i, rid_i, cnt_f, out);
  hipLaunchKernelGGL(region_tail_kernel, dim3(NB * (NS - NTOK)), dim3(192), 0, stream,
                     ew, out);
}
